// PLASALLM_26328149524506
// MI455X (gfx1250) — matrix operand order repaired and checked by reading, not compiled, not run on hardware
//
#include <hip/hip_runtime.h>
#include <hip/hip_bf16.h>
#include <cstdint>
#include <cstddef>

// ---------------------------------------------------------------------------
// Model constants (fixed by the reference)
// ---------------------------------------------------------------------------
constexpr int cV = 50257, cD = 1024, cH = 16, cL = 8, cF = 4096, cS = 1024;
constexpr int cHI = 4, cDI = 64, cTOPK = 512, cB = 2;
constexpr int cR = cB * cS;          // 2048 token rows
constexpr float cEPS = 1e-6f;

typedef __attribute__((ext_vector_type(16))) __bf16 v16bf;
typedef __attribute__((ext_vector_type(8)))  float  v8f;

union FragBF { uint4 q[2]; v16bf v; };   // 32B WMMA A/B fragment
union U4H    { uint4 q; __bf16 h[8]; };  // 16B <-> 8 bf16

// ---------------------------------------------------------------------------
// fp32 -> bf16 bulk convert (embedding table)
// ---------------------------------------------------------------------------
__global__ __launch_bounds__(256) void k_cvt_bf16(const float* __restrict__ in,
                                                  __bf16* __restrict__ out, long n) {
  long i = (long)blockIdx.x * blockDim.x + threadIdx.x;
  long stride = (long)gridDim.x * blockDim.x;
  for (; i < n; i += stride) out[i] = (__bf16)in[i];
}

// ---------------------------------------------------------------------------
// fp32 [Z][K][N] -> bf16 [Z][N][K] tiled transpose (weights, once per launch).
// Makes every GEMM B-operand K-major so staging is 16B-contiguous.
// ---------------------------------------------------------------------------
__global__ __launch_bounds__(256) void k_cvt_t(const float* __restrict__ in,
                                               __bf16* __restrict__ out,
                                               int K, int N) {
  __shared__ float t[32][33];
  const long zoff = (long)blockIdx.z * K * N;
  const int n0 = blockIdx.x * 32, k0 = blockIdx.y * 32;
  const int tx = threadIdx.x & 31, ty = threadIdx.x >> 5;   // 8 rows per pass
  #pragma unroll
  for (int j = 0; j < 4; ++j) {
    const int k = k0 + ty + j * 8, n = n0 + tx;
    t[ty + j * 8][tx] = (k < K && n < N) ? in[zoff + (long)k * N + n] : 0.f;
  }
  __syncthreads();
  #pragma unroll
  for (int j = 0; j < 4; ++j) {
    const int n = n0 + ty + j * 8, k = k0 + tx;
    if (n < N && k < K) out[zoff + (long)n * K + k] = (__bf16)t[tx][ty + j * 8];
  }
}

// ---------------------------------------------------------------------------
// x = embed_w[token] + pos_w[s]       (one block per token row)
// ---------------------------------------------------------------------------
__global__ __launch_bounds__(256) void k_embed(const int* __restrict__ tokens,
                                               const float* __restrict__ emb,
                                               const float* __restrict__ pos,
                                               float* __restrict__ x) {
  const long row = blockIdx.x;             // b*S + s
  const int  s   = (int)(row % cS);
  const long tok = tokens[row];
  const float* er = emb + tok * (long)cD;
  const float* pr = pos + (long)s * cD;
  float* xr = x + row * (long)cD;
  for (int d = threadIdx.x; d < cD; d += 256) xr[d] = er[d] + pr[d];
}

// ---------------------------------------------------------------------------
// RMSNorm: out_bf16 = w * x / (||x||*D^-1/2 + eps)   (one block per row)
// ---------------------------------------------------------------------------
__global__ __launch_bounds__(256) void k_rmsnorm(const float* __restrict__ x,
                                                 const float* __restrict__ w,
                                                 __bf16* __restrict__ out) {
  const long row = blockIdx.x;
  const float* xr = x + row * (long)cD;
  __shared__ float sm[256];
  float ss = 0.f;
  #pragma unroll
  for (int j = 0; j < cD / 256; ++j) {
    float v = xr[threadIdx.x + 256 * j];
    ss += v * v;
  }
  sm[threadIdx.x] = ss; __syncthreads();
  for (int s = 128; s > 0; s >>= 1) {
    if (threadIdx.x < s) sm[threadIdx.x] += sm[threadIdx.x + s];
    __syncthreads();
  }
  const float nrm = sqrtf(sm[0] * (1.f / cD));
  const float inv = 1.f / (nrm + cEPS);
  __bf16* orow = out + row * (long)cD;
  #pragma unroll
  for (int j = 0; j < cD / 256; ++j) {
    int d = threadIdx.x + 256 * j;
    orow[d] = (__bf16)(w[d] * xr[d] * inv);
  }
}

// ---------------------------------------------------------------------------
// bf16 WMMA GEMM:  C[M,N](f32 acc) = A[M,K] * B^T   with B stored [N,K] K-major
//   EPI: 0 = f32 | 1 = f32 + residual | 2 = bf16 | 3 = bf16(gelu)
//        4 = bf16 transposed per-batch V store: vT[b][col][s], b=m/S, s=m%S
//   grid.z batching: z1 = z/zdiv, z2 = z%zdiv; per-matrix offsets s?1*z1+s?2*z2
//
// Block: 128 threads (4 waves). Tile 128x64; wave = 32 rows x 64 cols,
// acc[2][4], 8 WMMAs per K=32 step off 4 shared B fragments.
// B staging: double-buffered LDS fed by global_load_async_to_lds_b128
// (ASYNCcnt), issued for tile i+1 before the WMMA burst on tile i.
// Ragged-N blocks fall back to guarded register staging (block-uniform branch).
// Requires: M % 128 == 0, K % 32 == 0 (true at every call site). N ragged OK.
// ---------------------------------------------------------------------------
template<int EPI>
__global__ __launch_bounds__(128) void gemm_bf16_kernel(
    const __bf16* __restrict__ A, int lda, long sA1, long sA2,
    const __bf16* __restrict__ Bm, int ldb, long sB1, long sB2,
    float* __restrict__ Cf, __bf16* __restrict__ Cb,
    const float* __restrict__ Res,
    int ldc, long sC1, long sC2,
    int N, int K, int zdiv) {
  // LDS B tiles (ping-pong), K-contiguous per column: Bs[buf][col][k].
  // Stride 56 (=112B = 7*16B): 16B-aligned b128 ds ops, 28-bank column spread.
  __shared__ __bf16 Bs[2][64][56];

  const int tid  = threadIdx.x;
  const int lane = tid & 31;
  const int wave = tid >> 5;
  const int z1 = blockIdx.z / zdiv, z2 = blockIdx.z % zdiv;
  A  += (long)z1 * sA1 + (long)z2 * sA2;
  Bm += (long)z1 * sB1 + (long)z2 * sB2;
  const long coff = (long)z1 * sC1 + (long)z2 * sC2;

  const int n0 = blockIdx.x * 64;
  const int mw = blockIdx.y * 128 + wave * 32;       // wave's first output row
  const bool fullN = (n0 + 64 <= N);                  // block-uniform (scalar branch)

  v8f acc[2][4];
  #pragma unroll
  for (int mi = 0; mi < 2; ++mi)
    #pragma unroll
    for (int t = 0; t < 4; ++t)
      #pragma unroll
      for (int r = 0; r < 8; ++r) acc[mi][t][r] = 0.f;

  const int rA  = lane & 15;          // A row within 16-tile / C column within tile
  const int khA = (lane >> 4) * 8;    // A K-half select
  const int khB = (lane >> 4) * 8;    // B K-half select

  // staging task geometry: two 16B chunks per thread (cols tc, tc+32; chunk tch)
  const int tc = tid >> 2, tch = tid & 3;

  U4H st0, st1;   // register staging (ragged-N fallback only)

  // ---- async stage: global -> LDS[buf], 16B per lane per op (ASYNCcnt) ----
  auto stage_async = [&](int k0, int buf) {
    const __bf16* g0 = Bm + (long)(n0 + tc)      * ldb + k0 + tch * 8;
    const __bf16* g1 = Bm + (long)(n0 + tc + 32) * ldb + k0 + tch * 8;
    unsigned l0 = (unsigned)(uintptr_t)&Bs[buf][tc][tch * 8];
    unsigned l1 = (unsigned)(uintptr_t)&Bs[buf][tc + 32][tch * 8];
    asm volatile("global_load_async_to_lds_b128 %0, %1, off"
                 :: "v"(l0), "v"(g0) : "memory");
    asm volatile("global_load_async_to_lds_b128 %0, %1, off"
                 :: "v"(l1), "v"(g1) : "memory");
  };
  // ---- guarded register stage (ragged N; single buffer 0) ----
  auto stage_load = [&](int k0) {
    st0.q.x = st0.q.y = st0.q.z = st0.q.w = 0u;
    st1.q.x = st1.q.y = st1.q.z = st1.q.w = 0u;
    if (n0 + tc < N)
      st0.q = *(const uint4*)(Bm + (long)(n0 + tc)      * ldb + k0 + tch * 8);
    if (n0 + tc + 32 < N)
      st1.q = *(const uint4*)(Bm + (long)(n0 + tc + 32) * ldb + k0 + tch * 8);
  };
  auto stage_write = [&]() {
    *(uint4*)&Bs[0][tc][tch * 8]      = st0.q;
    *(uint4*)&Bs[0][tc + 32][tch * 8] = st1.q;
  };

  int buf = 0;
  if (fullN) stage_async(0, 0); else stage_load(0);

  for (int k0 = 0; k0 < K; k0 += 32) {
    int rb;
    if (fullN) {
      asm volatile("s_wait_asynccnt 0x0" ::: "memory");  // this wave's copies done
      __syncthreads();                                    // ... all waves' copies done
      if (k0 + 32 < K) stage_async(k0 + 32, buf ^ 1);     // prefetch next tile
      rb = buf;
    } else {
      __syncthreads();
      stage_write();
      __syncthreads();
      if (k0 + 32 < K) stage_load(k0 + 32);
      rb = 0;
    }

    // A fragments: lane<16 holds K {0..7,16..23}, lane>=16 holds {8..15,24..31}
    FragBF fa[2];
    #pragma unroll
    for (int mi = 0; mi < 2; ++mi) {
      const __bf16* ap = A + (long)(mw + mi * 16 + rA) * lda + k0 + khA;
      fa[mi].q[0] = *(const uint4*)(ap);
      fa[mi].q[1] = *(const uint4*)(ap + 16);
      __builtin_prefetch(ap + 32, 0, 3);   // next A K-tile into near cache
    }
    // Preload all B fragments (distinct regs -> WMMAs issue back-to-back)
    FragBF fb[4];
    #pragma unroll
    for (int t = 0; t < 4; ++t) {
      const __bf16* bp = &Bs[rb][t * 16 + rA][khB];
      fb[t].q[0] = *(const uint4*)(bp);
      fb[t].q[1] = *(const uint4*)(bp + 16);
    }
    #pragma unroll
    for (int t = 0; t < 4; ++t)
      #pragma unroll
      for (int mi = 0; mi < 2; ++mi)
        acc[mi][t] = __builtin_amdgcn_wmma_f32_16x16x32_bf16(
            false, fa[mi].v, false, fb[t].v, (short)0, acc[mi][t], false, false);

    buf ^= 1;
  }

  // Epilogue. C layout: VGPR r, lane<16 -> row r, lane>=16 -> row r+8; col = lane&15.
  const int rowl = (lane >> 4) * 8;
  #pragma unroll
  for (int mi = 0; mi < 2; ++mi) {
    #pragma unroll
    for (int t = 0; t < 4; ++t) {
      const int col = n0 + t * 16 + (lane & 15);
      if (col < N) {
        #pragma unroll
        for (int r = 0; r < 8; ++r) {
          const int  m = mw + mi * 16 + rowl + r;
          const long idx = coff + (long)m * ldc + col;
          const float v = acc[mi][t][r];
          if (EPI == 0)      Cf[idx] = v;
          else if (EPI == 1) Cf[idx] = v + Res[idx];
          else if (EPI == 2) Cb[idx] = (__bf16)v;
          else if (EPI == 3) {     // exact GELU: 0.5x(1+erf(x/sqrt(2)))
            Cb[idx] = (__bf16)(0.5f * v * (1.f + erff(v * 0.70710678118654752f)));
          } else {                 // EPI 4: vT[b][col][s], b=m/S, s=m%S (K-major V)
            const long tidx = ((long)(m >> 10) * cD + col) * cS + (m & (cS - 1));
            Cb[tidx] = (__bf16)v;
          }
        }
      }
    }
  }
}

// ---------------------------------------------------------------------------
// idx_scores[b,q,k] = sum_h wt[b,q,h] * relu(idx_logits[b,h,q,k])
// ---------------------------------------------------------------------------
__global__ __launch_bounds__(256) void k_idx_combine(const float* __restrict__ lg,
                                                     const float* __restrict__ wt,
                                                     float* __restrict__ sc) {
  const long i = (long)blockIdx.x * 256 + threadIdx.x;   // over B*S*S
  const long b = i / ((long)cS * cS);
  const long rem = i % ((long)cS * cS);
  const long q = rem / cS, k = rem % cS;
  const float* wr = wt + ((long)b * cS + q) * cHI;
  float a = 0.f;
  #pragma unroll
  for (int h = 0; h < cHI; ++h) {
    const float l = lg[(((long)b * cHI + h) * cS + q) * cS + k];
    a += wr[h] * fmaxf(l, 0.f);
  }
  sc[i] = a;
}

// ---------------------------------------------------------------------------
// Top-K mask per (b,q): radix threshold search on order-preserving uint keys.
// allowed = causal & (score among top TOPK). Writes 32 mask words per row.
// ---------------------------------------------------------------------------
__device__ __forceinline__ unsigned fkey(float x) {
  unsigned u = __float_as_uint(x);
  return (u & 0x80000000u) ? ~u : (u | 0x80000000u);
}

__global__ __launch_bounds__(256) void k_topk_mask(const float* __restrict__ sc,
                                                   unsigned* __restrict__ mask) {
  const int tid = threadIdx.x;
  const int q = blockIdx.x % cS;
  const int b = blockIdx.x / cS;
  const float* row = sc + ((long)b * cS + q) * cS;
  const int cnt = q + 1;                       // causal candidates

  __shared__ int sred[256];
  __shared__ unsigned wrd[32];

  unsigned mykey[cS / 256];
  #pragma unroll
  for (int j = 0; j < cS / 256; ++j) {
    const int k = tid + 256 * j;
    mykey[j] = (k <= q) ? fkey(row[k]) : 0u;   // excluded -> key 0 (< any cand)
  }

  unsigned T = 0u;
  if (cnt > cTOPK) {
    for (int bit = 31; bit >= 0; --bit) {
      const unsigned cand = T | (1u << bit);
      int c = 0;
      #pragma unroll
      for (int j = 0; j < cS / 256; ++j) c += (mykey[j] >= cand);
      sred[tid] = c; __syncthreads();
      for (int s = 128; s > 0; s >>= 1) {
        if (tid < s) sred[tid] += sred[tid + s];
        __syncthreads();
      }
      const int tot = sred[0];
      __syncthreads();
      if (tot >= cTOPK) T = cand;              // largest T with count >= TOPK
    }
  }

  if (tid < 32) wrd[tid] = 0u;
  __syncthreads();
  #pragma unroll
  for (int j = 0; j < cS / 256; ++j) {
    const int k = tid + 256 * j;
    const bool ok = (k <= q) && (cnt <= cTOPK || mykey[j] >= T);
    if (ok) atomicOr(&wrd[k >> 5], 1u << (k & 31));
  }
  __syncthreads();
  if (tid < 32) mask[((long)b * cS + q) * 32 + tid] = wrd[tid];
}

// ---------------------------------------------------------------------------
// Masked softmax over attention logits, scale 1/sqrt(DH)=0.125, write bf16 P.
// One block per (b,h,q) row.
// ---------------------------------------------------------------------------
__global__ __launch_bounds__(256) void k_softmax(const float* __restrict__ att,
                                                 const unsigned* __restrict__ mask,
                                                 __bf16* __restrict__ p) {
  const int tid = threadIdx.x;
  const long row = blockIdx.x;                 // (b*H + h)*S + q
  const int q = (int)(row % cS);
  const int b = (int)(row / ((long)cH * cS));
  const float* ar = att + row * (long)cS;
  const unsigned* mw = mask + ((long)b * cS + q) * 32;
  __shared__ float sm[256];

  float vals[cS / 256];
  float lmax = -3.4e38f;
  #pragma unroll
  for (int j = 0; j < cS / 256; ++j) {
    const int k = tid + 256 * j;
    const bool ok = (mw[k >> 5] >> (k & 31)) & 1u;
    vals[j] = ok ? ar[k] * 0.125f : -3.4e38f;
    lmax = fmaxf(lmax, vals[j]);
  }
  sm[tid] = lmax; __syncthreads();
  for (int s = 128; s > 0; s >>= 1) {
    if (tid < s) sm[tid] = fmaxf(sm[tid], sm[tid + s]);
    __syncthreads();
  }
  const float M = sm[0]; __syncthreads();

  float lsum = 0.f;
  #pragma unroll
  for (int j = 0; j < cS / 256; ++j) {
    vals[j] = (vals[j] > -3.0e38f) ? __expf(vals[j] - M) : 0.f;
    lsum += vals[j];
  }
  sm[tid] = lsum; __syncthreads();
  for (int s = 128; s > 0; s >>= 1) {
    if (tid < s) sm[tid] += sm[tid + s];
    __syncthreads();
  }
  const float inv = 1.f / sm[0];
  __bf16* pr = p + row * (long)cS;
  #pragma unroll
  for (int j = 0; j < cS / 256; ++j) pr[tid + 256 * j] = (__bf16)(vals[j] * inv);
}

// ---------------------------------------------------------------------------
// Host-side launchers
// ---------------------------------------------------------------------------
template<int EPI>
static void gemm(hipStream_t st,
                 const __bf16* A, int lda, long sA1, long sA2,
                 const __bf16* B, int ldb, long sB1, long sB2,
                 float* Cf, __bf16* Cb, const float* Res,
                 int ldc, long sC1, long sC2,
                 int M, int N, int K, int Z, int zdiv) {
  dim3 g((unsigned)((N + 63) / 64), (unsigned)(M / 128), (unsigned)Z);
  gemm_bf16_kernel<EPI><<<g, 128, 0, st>>>(A, lda, sA1, sA2, B, ldb, sB1, sB2,
                                           Cf, Cb, Res, ldc, sC1, sC2, N, K, zdiv);
}

static void cvt(hipStream_t st, const float* in, __bf16* out, long n) {
  long blocks = (n + 255) / 256; if (blocks > 8192) blocks = 8192;
  k_cvt_bf16<<<(unsigned)blocks, 256, 0, st>>>(in, out, n);
}

static void cvt_t(hipStream_t st, const float* in, __bf16* out, int K, int N, int Z) {
  dim3 g((unsigned)((N + 31) / 32), (unsigned)((K + 31) / 32), (unsigned)Z);
  k_cvt_t<<<g, 256, 0, st>>>(in, out, K, N);
}

// ---------------------------------------------------------------------------
// kernel_launch
// ---------------------------------------------------------------------------
extern "C" void kernel_launch(void* const* d_in, const int* in_sizes, int n_in,
                              void* d_out, int out_size, void* d_ws, size_t ws_size,
                              hipStream_t stream) {
  (void)in_sizes; (void)n_in; (void)out_size; (void)ws_size;

  const int*   tokens = (const int*)  d_in[0];
  const float* emb_w  = (const float*)d_in[1];
  const float* pos_w  = (const float*)d_in[2];
  const float* wq     = (const float*)d_in[3];
  const float* wk     = (const float*)d_in[4];
  const float* wv     = (const float*)d_in[5];
  const float* wo     = (const float*)d_in[6];
  const float* iq     = (const float*)d_in[7];
  const float* ik     = (const float*)d_in[8];
  const float* iw     = (const float*)d_in[9];
  const float* w1     = (const float*)d_in[10];
  const float* w2     = (const float*)d_in[11];
  const float* norm1  = (const float*)d_in[12];
  const float* norm2  = (const float*)d_in[13];
  const float* normf  = (const float*)d_in[14];
  float* logits       = (float*)d_out;

  // ---- workspace bump allocator ----
  char* wp = (char*)d_ws;
  auto alloc = [&](size_t bytes) -> void* {
    void* r = (void*)wp;
    wp += (bytes + 255) & ~(size_t)255;
    return r;
  };
  const long DD  = (long)cD * cD;
  const long DHI = (long)cD * cHI * cDI;   // iq per layer
  // bf16 weights, all stored K-major ([N][K], K = input dim)
  __bf16* wq_bf = (__bf16*)alloc(2L * cL * DD);
  __bf16* wk_bf = (__bf16*)alloc(2L * cL * DD);
  __bf16* wv_bf = (__bf16*)alloc(2L * cL * DD);
  __bf16* wo_bf = (__bf16*)alloc(2L * cL * DD);
  __bf16* iq_bf = (__bf16*)alloc(2L * cL * DHI);
  __bf16* ik_bf = (__bf16*)alloc(2L * cL * cD * cDI);
  __bf16* iw_bf = (__bf16*)alloc(2L * cL * cD * cHI);
  __bf16* w1_bf = (__bf16*)alloc(2L * cL * cD * cF);
  __bf16* w2_bf = (__bf16*)alloc(2L * cL * cF * cD);
  __bf16* em_bf = (__bf16*)alloc(2L * cV * cD);     // [V][D] is already K-major
  // bf16 activations
  __bf16* h_bf    = (__bf16*)alloc(2L * cR * cD);
  __bf16* q_bf    = (__bf16*)alloc(2L * cR * cD);
  __bf16* k_bf    = (__bf16*)alloc(2L * cR * cD);
  __bf16* vT_bf   = (__bf16*)alloc(2L * cB * cD * cS);  // [B][D][S] K-major V
  __bf16* qi_bf   = (__bf16*)alloc(2L * cR * cHI * cDI);
  __bf16* ki_bf   = (__bf16*)alloc(2L * cR * cDI);
  __bf16* attn_bf = (__bf16*)alloc(2L * cR * cD);
  __bf16* u_bf    = (__bf16*)alloc(2L * cR * cF);
  __bf16* p_bf    = (__bf16*)alloc(2L * cB * cH * cS * cS);
  // f32 buffers
  float* x      = (float*)alloc(4L * cR * cD);
  float* wt     = (float*)alloc(4L * cR * cHI);
  float* idxlog = (float*)alloc(4L * cB * cHI * cS * cS);
  float* idxsc  = (float*)alloc(4L * cB * cS * cS);
  float* att    = (float*)alloc(4L * cB * cH * cS * cS);
  unsigned* msk = (unsigned*)alloc(4L * cB * cS * 32);

  // ---- convert weights to bf16, transposed to K-major (once per launch) ----
  cvt_t(stream, wq, wq_bf, cD, cD, cL);
  cvt_t(stream, wk, wk_bf, cD, cD, cL);
  cvt_t(stream, wv, wv_bf, cD, cD, cL);
  cvt_t(stream, wo, wo_bf, cD, cD, cL);
  cvt_t(stream, iq, iq_bf, cD, cHI * cDI, cL);
  cvt_t(stream, ik, ik_bf, cD, cDI, cL);
  cvt_t(stream, iw, iw_bf, cD, cHI, cL);
  cvt_t(stream, w1, w1_bf, cD, cF, cL);
  cvt_t(stream, w2, w2_bf, cF, cD, cL);
  cvt(stream, emb_w, em_bf, (long)cV * cD);

  // ---- embedding ----
  k_embed<<<cR, 256, 0, stream>>>(tokens, emb_w, pos_w, x);

  const long SS = (long)cS * cS;
  // ---- transformer layers ----
  for (int l = 0; l < cL; ++l) {
    const __bf16* Wq = wq_bf + (long)l * DD;
    const __bf16* Wk = wk_bf + (long)l * DD;
    const __bf16* Wv = wv_bf + (long)l * DD;
    const __bf16* Wo = wo_bf + (long)l * DD;
    const __bf16* Iq = iq_bf + (long)l * DHI;
    const __bf16* Ik = ik_bf + (long)l * cD * cDI;
    const __bf16* Iw = iw_bf + (long)l * cD * cHI;
    const __bf16* W1 = w1_bf + (long)l * cD * cF;
    const __bf16* W2 = w2_bf + (long)l * cF * cD;
    const float* n1 = norm1 + (long)l * cD;
    const float* n2 = norm2 + (long)l * cD;

    // h = rmsnorm(x, n1)
    k_rmsnorm<<<cR, 256, 0, stream>>>(x, n1, h_bf);

    // q/k = h @ W{q,k} -> bf16 [R,D];  v -> vT_bf [B][D][S] (EPI 4)
    gemm<2>(stream, h_bf, cD,0,0, Wq, cD,0,0, nullptr, q_bf, nullptr, cD,0,0, cR, cD, cD, 1, 1);
    gemm<2>(stream, h_bf, cD,0,0, Wk, cD,0,0, nullptr, k_bf, nullptr, cD,0,0, cR, cD, cD, 1, 1);
    gemm<4>(stream, h_bf, cD,0,0, Wv, cD,0,0, nullptr, vT_bf, nullptr, cD,0,0, cR, cD, cD, 1, 1);
    // indexer projections
    gemm<2>(stream, h_bf, cD,0,0, Iq, cD,0,0, nullptr, qi_bf, nullptr, cHI*cDI,0,0, cR, cHI*cDI, cD, 1, 1);
    gemm<2>(stream, h_bf, cD,0,0, Ik, cD,0,0, nullptr, ki_bf, nullptr, cDI,0,0, cR, cDI, cD, 1, 1);
    gemm<0>(stream, h_bf, cD,0,0, Iw, cD,0,0, wt, nullptr, nullptr, cHI,0,0, cR, cHI, cD, 1, 1);

    // indexer logits[b,h] = qi_h [S,DI] @ ki^T   (z = b*HI + h)
    gemm<0>(stream,
            qi_bf, cHI*cDI, (long)cS*cHI*cDI, cDI,
            ki_bf, cDI,     (long)cS*cDI,     0,
            idxlog, nullptr, nullptr, cS, (long)cHI*SS, SS,
            cS, cS, cDI, cB*cHI, cHI);
    k_idx_combine<<<(unsigned)((cB*SS)/256), 256, 0, stream>>>(idxlog, wt, idxsc);
    k_topk_mask<<<cB*cS, 256, 0, stream>>>(idxsc, msk);

    // attention logits[b,h] = q_h [S,DH] @ k_h^T  (z = b*H + h)
    gemm<0>(stream,
            q_bf, cD, (long)cS*cD, cDI,
            k_bf, cD, (long)cS*cD, cDI,
            att, nullptr, nullptr, cS, (long)cH*SS, SS,
            cS, cS, cDI, cB*cH, cH);
    k_softmax<<<cB*cH*cS, 256, 0, stream>>>(att, msk, p_bf);

    // out_h = P [S,S] @ v_h  (v K-major: vT[b] is [D][S])  -> attn_bf, ldc=D
    gemm<2>(stream,
            p_bf, cS, (long)cH*SS, SS,
            vT_bf, cS, (long)cD*cS, (long)cDI*cS,
            nullptr, attn_bf, nullptr, cD, (long)cS*cD, cDI,
            cS, cDI, cS, cB*cH, cH);

    // x = attn @ Wo + x
    gemm<1>(stream, attn_bf, cD,0,0, Wo, cD,0,0, x, nullptr, x, cD,0,0, cR, cD, cD, 1, 1);

    // FFN: x = gelu(rmsnorm(x) @ W1) @ W2 + x
    k_rmsnorm<<<cR, 256, 0, stream>>>(x, n2, h_bf);
    gemm<3>(stream, h_bf, cD,0,0, W1, cD,0,0, nullptr, u_bf, nullptr, cF,0,0, cR, cF, cD, 1, 1);
    gemm<1>(stream, u_bf, cF,0,0, W2, cF,0,0, x, nullptr, x, cD,0,0, cR, cD, cF, 1, 1);
  }

  // ---- final norm + tied LM head: logits = rmsnorm(x) @ embed^T ----
  k_rmsnorm<<<cR, 256, 0, stream>>>(x, normf, h_bf);
  gemm<0>(stream, h_bf, cD,0,0, em_bf, cD,0,0,
          logits, nullptr, nullptr, cV,0,0, cR, cV, cD, 1, 1);
}
